// AttentionPairBias_63668595195923
// MI455X (gfx1250) — matrix operand order repaired and checked by reading, not compiled, not run on hardware
//
#include <hip/hip_runtime.h>

// ---------------------------------------------------------------------------
// AttentionPairBias on gfx1250 (CDNA5). bf16 WMMA everywhere, flash-style
// online softmax streaming the 268MB pair_logits tensor once (HBM roofline:
// ~11.5us at 23.3 TB/s; total math ~9.5 GFLOP is far below that).
// Softmax denominator is computed BY the P*V WMMA via an all-ones pad column
// in V^T (d=24), so the inner loop has no sum reduction -- only the max.
// ---------------------------------------------------------------------------

#define N_TOK 2048
#define CDIM  384
#define HEADS 16
#define DHEAD 24
#define DPAD  32
#define HSTRIDE (HEADS * DPAD)   // 512 padded channels per token

typedef __attribute__((ext_vector_type(16))) __bf16 v16bf;
typedef __attribute__((ext_vector_type(8)))  __bf16 v8bf;
typedef __attribute__((ext_vector_type(8)))  float  v8f;

__device__ inline __bf16 f32_to_bf16(float f) {
  unsigned u = __builtin_bit_cast(unsigned, f);
  unsigned r = u + 0x7FFFu + ((u >> 16) & 1u);   // round-to-nearest-even
  unsigned short h = (unsigned short)(r >> 16);
  return __builtin_bit_cast(__bf16, h);
}
__device__ inline float bf16_to_f32(__bf16 b) {
  unsigned short h = __builtin_bit_cast(unsigned short, b);
  unsigned u = ((unsigned)h) << 16;
  return __builtin_bit_cast(float, u);
}

// Two contiguous 16B (8 x bf16) loads -> one v16bf fragment.
__device__ inline v16bf ld_frag(const __bf16* p0, const __bf16* p1) {
  v8bf a = *(const v8bf*)p0;
  v8bf b = *(const v8bf*)p1;
  return __builtin_shufflevector(a, b, 0, 1, 2, 3, 4, 5, 6, 7,
                                 8, 9, 10, 11, 12, 13, 14, 15);
}

// ---------------------------------------------------------------------------
// LayerNorm: one block (128 thr) per row, write bf16 xn
// ---------------------------------------------------------------------------
__global__ __launch_bounds__(128) void ln_kernel(
    const float* __restrict__ x, const float* __restrict__ w,
    const float* __restrict__ b, __bf16* __restrict__ xn)
{
  __shared__ float red[8];
  const int row = blockIdx.x;
  const float* xr = x + (size_t)row * CDIM;
  float s = 0.f, s2 = 0.f;
  for (int c = threadIdx.x; c < CDIM; c += 128) { float v = xr[c]; s += v; s2 += v * v; }
  for (int off = 1; off < 32; off <<= 1) {
    s  += __shfl_xor(s,  off, 32);
    s2 += __shfl_xor(s2, off, 32);
  }
  const int wave = threadIdx.x >> 5, lane = threadIdx.x & 31;
  if (lane == 0) { red[wave] = s; red[4 + wave] = s2; }
  __syncthreads();
  const float ts  = red[0] + red[1] + red[2] + red[3];
  const float ts2 = red[4] + red[5] + red[6] + red[7];
  const float mu  = ts / CDIM;
  const float var = ts2 / CDIM - mu * mu;
  const float rinv = rsqrtf(var + 1e-5f);
  for (int c = threadIdx.x; c < CDIM; c += 128) {
    float v = (xr[c] - mu) * rinv * w[c] + b[c];
    xn[(size_t)row * CDIM + c] = f32_to_bf16(v);
  }
}

// ---------------------------------------------------------------------------
// fp32 -> bf16 conversion / padded-buffer fills
// ---------------------------------------------------------------------------
__global__ __launch_bounds__(256) void f2bf_kernel(
    const float* __restrict__ src, __bf16* __restrict__ dst, int n)
{
  int i = blockIdx.x * 256 + threadIdx.x;
  if (i < n) dst[i] = f32_to_bf16(src[i]);
}

__global__ __launch_bounds__(256) void zero_bf16_kernel(__bf16* dst, int n)
{
  int i = blockIdx.x * 256 + threadIdx.x;
  if (i < n) dst[i] = f32_to_bf16(0.f);
}

// V^T pad fill: 1.0 on the d==24 row (softmax-denominator ones column),
// 0 elsewhere. The V projection overwrites d<24; pad rows survive.
__global__ __launch_bounds__(256) void fill_vt_kernel(__bf16* dst, int n)
{
  int i = blockIdx.x * 256 + threadIdx.x;
  if (i < n) {
    const int d = (i / N_TOK) % DPAD;
    dst[i] = f32_to_bf16(d == DHEAD ? 1.f : 0.f);
  }
}

// ---------------------------------------------------------------------------
// Y = A(bf16, MxK=384) @ W.T (+bias)*scale ; W stored [Nout x K] bf16.
// One 16x16 tile per wave, 12 WMMA k-steps. Output modes:
//   0: plain bf16  Y[row*CDIM + o]
//   1: head-padded bf16  Y[row*HSTRIDE + (o/24)*32 + o%24]     (Q, K)
//   2: transposed head-padded bf16 Y[((o/24)*32 + o%24)*N_TOK + row]  (V)
//   3: fp32  Y[row*CDIM + o]                                   (final out)
// Fragment layouts used here (A and B share contiguous k order):
//   A 16x32: lane holds row L%16; elems 0..7 -> k=16*hi+(0..7),
//            elems 8..15 -> k=16*hi+8+(0..7)     (two b128 loads)
//   B 32x16: lane holds col L%16; elems -> k=16*hi+(0..15) (two b128 loads)
//   C/D 16x16 f32: elem r -> row r+8*hi, col L%16
// ---------------------------------------------------------------------------
template <int MODE>
__global__ __launch_bounds__(128) void gemm_xwT_kernel(
    const __bf16* __restrict__ A, const __bf16* __restrict__ Wb,
    const float* __restrict__ bias, float scale,
    __bf16* __restrict__ Ybf, float* __restrict__ Yf)
{
  const int lane = threadIdx.x & 31;
  const int wave = threadIdx.x >> 5;
  const int col  = lane & 15;
  const int hi   = lane >> 4;
  const int m0   = blockIdx.x * 16;
  const int n0   = (blockIdx.y * 4 + wave) * 16;

  v8f acc = {};
  const __bf16* arow = A + (size_t)(m0 + col) * CDIM;
  const __bf16* brow = Wb + (size_t)(n0 + col) * CDIM + 16 * hi;
  for (int k0 = 0; k0 < CDIM; k0 += 32) {
    v16bf a = ld_frag(arow + k0 + 16 * hi, arow + k0 + 16 * hi + 8);
    v16bf b = ld_frag(brow + k0, brow + k0 + 8);
    acc = __builtin_amdgcn_wmma_f32_16x16x32_bf16(false, a, false, b,
                                                  (short)0, acc, false, false);
  }
  const int o = n0 + col;
  const float bi = bias ? bias[o] : 0.f;
  const int pidx = (o / DHEAD) * DPAD + (o % DHEAD);
  for (int r = 0; r < 8; ++r) {
    const int row = m0 + r + 8 * hi;
    const float y = (acc[r] + bi) * scale;
    if (MODE == 0)      Ybf[(size_t)row * CDIM + o] = f32_to_bf16(y);
    else if (MODE == 1) Ybf[(size_t)row * HSTRIDE + pidx] = f32_to_bf16(y);
    else if (MODE == 2) Ybf[(size_t)pidx * N_TOK + row] = f32_to_bf16(y);
    else                Yf[(size_t)row * CDIM + o] = y;
  }
}

// ---------------------------------------------------------------------------
// Flash attention with pair bias. Block = 4 waves; each wave owns 16 queries
// of head blockIdx.y; 64 k-steps of 32 keys. Q/K head-padded [N][H][32],
// V transposed [H*32][N] with ones at d=24 (denominator column).
// ---------------------------------------------------------------------------
__global__ __launch_bounds__(128) void attn_kernel(
    const __bf16* __restrict__ Q, const __bf16* __restrict__ K,
    const __bf16* __restrict__ Vt, const float* __restrict__ pair,
    const float* __restrict__ mask, float* __restrict__ wa)
{
  __shared__ __bf16 plds[4][16 * 40];     // per-wave P tile, 80B rows (aligned)
  const int lane = threadIdx.x & 31;
  const int wave = threadIdx.x >> 5;
  const int col  = lane & 15;
  const int hi   = lane >> 4;
  const int h    = blockIdx.y;
  const int q0   = blockIdx.x * 64 + wave * 16;

  // Q A-fragment (zero-padded d in buffer), loaded once
  const __bf16* qbase = Q + (size_t)(q0 + col) * HSTRIDE + h * DPAD;
  const v16bf aq = ld_frag(qbase + 16 * hi, qbase + 16 * hi + 8);

  v8f acc0 = {}, acc1 = {};
  float mrow[8];
  for (int r = 0; r < 8; ++r) mrow[r] = -1e30f;

  const float* pairh = pair + (size_t)h * N_TOK * N_TOK;
  __bf16* pw = plds[wave];
  const __bf16* vrow0 = Vt + (size_t)(h * DPAD + col) * N_TOK + 16 * hi;
  const __bf16* vrow1 = Vt + (size_t)(h * DPAD + 16 + col) * N_TOK + 16 * hi;

  for (int kb = 0; kb < N_TOK; kb += 32) {
    // K B-fragments: K-dim = d (padded 32), cols = key indices
    const __bf16* kb0 = K + (size_t)(kb + col) * HSTRIDE + h * DPAD + 16 * hi;
    const __bf16* kb1 = kb0 + 16 * (size_t)HSTRIDE;
    v16bf bk0 = ld_frag(kb0, kb0 + 8);
    v16bf bk1 = ld_frag(kb1, kb1 + 8);
    const v8f z8 = {};
    v8f s0 = __builtin_amdgcn_wmma_f32_16x16x32_bf16(false, aq, false, bk0,
                                                     (short)0, z8, false, false);
    v8f s1 = __builtin_amdgcn_wmma_f32_16x16x32_bf16(false, aq, false, bk1,
                                                     (short)0, z8, false, false);

    // add mask bias + streamed pair_logits. Unconditional speculative prefetch
    // (TH=0: non-faulting, dropped on translation miss) 1KB ahead keeps the
    // 268MB stream flowing; no branches in the hot loop.
    const int kc0 = kb + col, kc1 = kb + 16 + col;
    const float mb0 = 1e9f * (mask[kc0] - 1.0f);
    const float mb1 = 1e9f * (mask[kc1] - 1.0f);
    for (int r = 0; r < 8; ++r) {
      const size_t prow = (size_t)(q0 + r + 8 * hi) * N_TOK;
      __builtin_prefetch(pairh + prow + kc0 + 256, 0, 0);
      s0[r] += pairh[prow + kc0] + mb0;
      s1[r] += pairh[prow + kc1] + mb1;
    }

    // online max (only reduction left; denominator comes from the ones col)
    for (int r = 0; r < 8; ++r) {
      float m = fmaxf(s0[r], s1[r]);
      for (int off = 1; off < 16; off <<= 1) m = fmaxf(m, __shfl_xor(m, off, 32));
      const float mnew = fmaxf(mrow[r], m);
      const float sc = __expf(mrow[r] - mnew);
      mrow[r] = mnew;
      acc0[r] *= sc; acc1[r] *= sc;
      s0[r] = __expf(s0[r] - mnew);
      s1[r] = __expf(s1[r] - mnew);
    }

    // re-stripe P (C-layout -> A-layout) through LDS
    for (int r = 0; r < 8; ++r) {
      const int row = r + 8 * hi;
      pw[row * 40 + col]      = f32_to_bf16(s0[r]);
      pw[row * 40 + 16 + col] = f32_to_bf16(s1[r]);
    }
    asm volatile("s_wait_dscnt 0" ::: "memory");

    const __bf16* prow = pw + col * 40;
    v16bf ap = ld_frag(prow + 16 * hi, prow + 16 * hi + 8);

    // V B-fragments from transposed V: contiguous along keys.
    // Column d=24 of bv1 is all ones -> acc1 col 8 accumulates sum(P).
    v16bf bv0 = ld_frag(vrow0 + kb, vrow0 + kb + 8);
    v16bf bv1 = ld_frag(vrow1 + kb, vrow1 + kb + 8);
    acc0 = __builtin_amdgcn_wmma_f32_16x16x32_bf16(false, ap, false, bv0,
                                                   (short)0, acc0, false, false);
    acc1 = __builtin_amdgcn_wmma_f32_16x16x32_bf16(false, ap, false, bv1,
                                                   (short)0, acc1, false, false);
  }

  // denominator: acc1 col 8 (d=24 ones column), broadcast from lane 16*hi+8
  for (int r = 0; r < 8; ++r) {
    const float l = __shfl(acc1[r], 16 * hi + 8, 32);
    const float inv = 1.0f / l;
    const int qrow = q0 + r + 8 * hi;
    wa[(size_t)qrow * CDIM + h * DHEAD + col] = acc0[r] * inv;
    if (col < DHEAD - 16)
      wa[(size_t)qrow * CDIM + h * DHEAD + 16 + col] = acc1[r] * inv;
  }
}

// ---------------------------------------------------------------------------
// gated = wa * sigmoid(G), written bf16 for the final WMMA GEMM
// ---------------------------------------------------------------------------
__global__ __launch_bounds__(256) void gate_kernel(
    const float* __restrict__ wa, const __bf16* __restrict__ g,
    __bf16* __restrict__ gated, int n)
{
  int i = blockIdx.x * 256 + threadIdx.x;
  if (i < n) {
    const float gv = bf16_to_f32(g[i]);
    const float sg = 1.f / (1.f + __expf(-gv));
    gated[i] = f32_to_bf16(wa[i] * sg);
  }
}

// ---------------------------------------------------------------------------
extern "C" void kernel_launch(void* const* d_in, const int* in_sizes, int n_in,
                              void* d_out, int out_size, void* d_ws, size_t ws_size,
                              hipStream_t stream) {
  (void)in_sizes; (void)n_in; (void)out_size; (void)ws_size;
  const float* x    = (const float*)d_in[0];
  const float* mask = (const float*)d_in[1];
  const float* pair = (const float*)d_in[2];
  const float* lnw  = (const float*)d_in[3];
  const float* lnb  = (const float*)d_in[4];
  const float* wq   = (const float*)d_in[5];
  const float* bq   = (const float*)d_in[6];
  const float* wk   = (const float*)d_in[7];
  const float* wv   = (const float*)d_in[8];
  const float* wg   = (const float*)d_in[9];
  const float* wo   = (const float*)d_in[10];
  float* out = (float*)d_out;

  const size_t NC  = (size_t)N_TOK * CDIM;     // 786432
  const size_t NCP = (size_t)N_TOK * HSTRIDE;  // 1048576 (padded)
  const size_t CC  = (size_t)CDIM * CDIM;      // 147456

  char* p = (char*)d_ws;
  __bf16* xn    = (__bf16*)p; p += NC * 2;
  __bf16* gb    = (__bf16*)p; p += NC * 2;
  __bf16* gated = (__bf16*)p; p += NC * 2;
  __bf16* qp    = (__bf16*)p; p += NCP * 2;    // [N][H][32], d padded
  __bf16* kp    = (__bf16*)p; p += NCP * 2;    // [N][H][32], d padded
  __bf16* vt    = (__bf16*)p; p += NCP * 2;    // [H*32][N], d padded (+ones row)
  __bf16* wqb   = (__bf16*)p; p += CC * 2;
  __bf16* wkb   = (__bf16*)p; p += CC * 2;
  __bf16* wvb   = (__bf16*)p; p += CC * 2;
  __bf16* wgb   = (__bf16*)p; p += CC * 2;
  __bf16* wob   = (__bf16*)p; p += CC * 2;
  float*  wabuf = (float*)p;  p += NC * 4;

  // 1. weights fp32 -> bf16; fill the padded operand buffers
  const int cvb = (int)((CC + 255) / 256);
  f2bf_kernel<<<cvb, 256, 0, stream>>>(wq, wqb, (int)CC);
  f2bf_kernel<<<cvb, 256, 0, stream>>>(wk, wkb, (int)CC);
  f2bf_kernel<<<cvb, 256, 0, stream>>>(wv, wvb, (int)CC);
  f2bf_kernel<<<cvb, 256, 0, stream>>>(wg, wgb, (int)CC);
  f2bf_kernel<<<cvb, 256, 0, stream>>>(wo, wob, (int)CC);
  const int zb = (int)((NCP + 255) / 256);
  zero_bf16_kernel<<<zb, 256, 0, stream>>>(qp, (int)NCP);
  zero_bf16_kernel<<<zb, 256, 0, stream>>>(kp, (int)NCP);
  fill_vt_kernel<<<zb, 256, 0, stream>>>(vt, (int)NCP);

  // 2. LayerNorm
  ln_kernel<<<N_TOK, 128, 0, stream>>>(x, lnw, lnb, xn);

  // 3. Projections (Q folds bias + D^-0.5; V written transposed)
  const dim3 ggrid(N_TOK / 16, CDIM / 64);
  const float qscale = 1.0f / sqrtf((float)DHEAD);
  gemm_xwT_kernel<1><<<ggrid, 128, 0, stream>>>(xn, wqb, bq, qscale, qp, nullptr);
  gemm_xwT_kernel<1><<<ggrid, 128, 0, stream>>>(xn, wkb, nullptr, 1.f, kp, nullptr);
  gemm_xwT_kernel<2><<<ggrid, 128, 0, stream>>>(xn, wvb, nullptr, 1.f, vt, nullptr);
  gemm_xwT_kernel<0><<<ggrid, 128, 0, stream>>>(xn, wgb, nullptr, 1.f, gb, nullptr);

  // 4. Flash attention with pair bias (streams the 268MB pair tensor once)
  attn_kernel<<<dim3(N_TOK / 64, HEADS), 128, 0, stream>>>(qp, kp, vt, pair, mask, wabuf);

  // 5. Gating
  gate_kernel<<<(int)((NC + 255) / 256), 256, 0, stream>>>(wabuf, gb, gated, (int)NC);

  // 6. Output projection -> fp32 d_out
  gemm_xwT_kernel<3><<<ggrid, 128, 0, stream>>>(gated, wob, nullptr, 1.f, nullptr, out);
}
